// Model03_56324201119855
// MI455X (gfx1250) — matrix operand order repaired and checked by reading, not compiled, not run on hardware
//
#include <hip/hip_runtime.h>

// ---------------------------------------------------------------------------
// Two GRUs (runoff: in=4,h=16 ; pre: in=20,h=1) over T=365, B=4096, then
// sigmoid(linear([h_runoff, out_pre_last16])).
//
// runoff GRU recurrence via v_wmma_f32_16x16x32_f16:
//   G[gate][batch] = W[gate][k] * [h;x][k][batch],  K = 16(h)+4(x), pad to 32.
// Gates-in-M / batch-in-N orientation lets the f32 C/D tile feed the next
// step's f16 B-matrix with just a half-wave swap (v_permlanex16) — no
// transpose. Activations use v_rcp_f32 instead of IEEE division to shorten
// the serial critical path.
// ---------------------------------------------------------------------------

typedef __attribute__((ext_vector_type(16))) _Float16 v16h;
typedef __attribute__((ext_vector_type(2)))  _Float16 h2v;
typedef __attribute__((ext_vector_type(8)))  float    v8f;
typedef __attribute__((ext_vector_type(8)))  int      v8i;

#define T_STEPS 365
#define B_TOT   4096
#define H_RUNV  16

__device__ __forceinline__ float fast_rcp(float x) {
    return __builtin_amdgcn_rcpf(x);                 // v_rcp_f32
}
__device__ __forceinline__ float sigf(float v) {
    return fast_rcp(1.0f + __expf(-v));              // v_exp + v_rcp
}
__device__ __forceinline__ float tanhf_fast(float v) {
    // tanh(x) = 1 - 2/(1 + e^{2x}); saturates correctly at +-inf
    return fmaf(-2.0f, fast_rcp(1.0f + __expf(2.0f * v)), 1.0f);
}
// lane i <-> lane i^16 swap, pure VALU (no DS round-trip / s_wait_dscnt)
__device__ __forceinline__ int xswap16(int v) {
    return __builtin_amdgcn_permlanex16(v, v, 0x76543210u, 0xfedcba98u, false, false);
}

// ---------------------------------------------------------------------------
// Kernel 1: runoff GRU, WMMA recurrence. One wave per 16-sample batch tile.
// ---------------------------------------------------------------------------
__global__ __launch_bounds__(256)
void gru_runoff_wmma(const float* __restrict__ runoff,   // [B, T, 4]
                     const float* __restrict__ Wih_r,    // [48, 4]
                     const float* __restrict__ Whh_r,    // [48, 16]
                     const float* __restrict__ bih_r,    // [48]
                     const float* __restrict__ bhh_r,    // [48]
                     float* __restrict__ hT_out) {       // [B, 16]
    const int  lane = threadIdx.x & 31;
    const int  wave = threadIdx.x >> 5;
    const int  base = (blockIdx.x * 8 + wave) * 16;   // batch tile start
    const int  col  = lane & 15;                      // A: gate row; B/C/D: batch col
    const bool lo   = (lane < 16);
    const int  kOff = lo ? 0 : 8;                     // C/D M-offset (A K-offset is 2*kOff)

    // ---- Loop-invariant A-tiles (contiguous K order, K = 16*half + s) -----------------
    // lanes 0-15 : element s -> K=s      (K=0..15)
    // lanes16-31 : element s -> K=16+s   (K=16..31)
    // K 0..15 = hidden (Whh), K 16..19 = input (Wih), K 20..31 = zero pad.
    v16h A_r, A_z, A_nh, A_nx;
    #pragma unroll
    for (int s = 0; s < 16; ++s) {
        const int K = s + 2 * kOff;
        float r = 0.f, z = 0.f, nh = 0.f, nx = 0.f;
        if (K < 16) {
            r  = Whh_r[(0  + col) * 16 + K];
            z  = Whh_r[(16 + col) * 16 + K];
            nh = Whh_r[(32 + col) * 16 + K];
        } else if (K < 20) {
            r  = Wih_r[(0  + col) * 4 + (K - 16)];
            z  = Wih_r[(16 + col) * 4 + (K - 16)];
            nx = Wih_r[(32 + col) * 4 + (K - 16)];
        }
        A_r[s]  = (_Float16)r;
        A_z[s]  = (_Float16)z;
        A_nh[s] = (_Float16)nh;
        A_nx[s] = (_Float16)nx;
    }

    // ---- Per-lane biases (gate index = j + kOff within each 16-gate chunk) --
    float br[8], bz[8], bxn[8], bhn[8];
    #pragma unroll
    for (int j = 0; j < 8; ++j) {
        const int g = j + kOff;
        br[j]  = bih_r[g]      + bhh_r[g];        // r gate: full bias fused
        bz[j]  = bih_r[16 + g] + bhh_r[16 + g];   // z gate: full bias fused
        bxn[j] = bih_r[32 + g];                   // n gate: x-side bias
        bhn[j] = bhh_r[32 + g];                   // n gate: h-side bias (scaled by r)
    }

    // ---- Streaming input, depth-2 register pipeline + L2 prefetch -----------
    const float* xrow = runoff + (size_t)(base + col) * (T_STEPS * 4);
    float4 xq0 = *(const float4*)(xrow);
    float4 xq1 = *(const float4*)(xrow + 4);

    v8f h = {0.f, 0.f, 0.f, 0.f, 0.f, 0.f, 0.f, 0.f};
    const v8f zc = {0.f, 0.f, 0.f, 0.f, 0.f, 0.f, 0.f, 0.f};

    for (int t = 0; t < T_STEPS; ++t) {
        const float4 xc = xq0;
        xq0 = xq1;
        const int tn = (t + 2 < T_STEPS) ? (t + 2) : (T_STEPS - 1);
        xq1 = *(const float4*)(xrow + tn * 4);
        const int tp = (t + 24 < T_STEPS) ? (t + 24) : (T_STEPS - 1);
        __builtin_prefetch(xrow + tp * 4, 0, 1);   // global_prefetch_b8, warm L2

        // Pack h -> 4 dwords of f16 pairs, then half-wave swap packed dwords.
        h2v p0 = {(_Float16)h[0], (_Float16)h[1]};
        h2v p1 = {(_Float16)h[2], (_Float16)h[3]};
        h2v p2 = {(_Float16)h[4], (_Float16)h[5]};
        h2v p3 = {(_Float16)h[6], (_Float16)h[7]};
        const int i0 = __builtin_bit_cast(int, p0);
        const int i1 = __builtin_bit_cast(int, p1);
        const int i2 = __builtin_bit_cast(int, p2);
        const int i3 = __builtin_bit_cast(int, p3);
        const int o0 = xswap16(i0);   // other half's hidden rows
        const int o1 = xswap16(i1);
        const int o2 = xswap16(i2);
        const int o3 = xswap16(i3);

        h2v xa = {(_Float16)xc.x, (_Float16)xc.y};
        h2v xb = {(_Float16)xc.z, (_Float16)xc.w};
        const int xi0 = __builtin_bit_cast(int, xa);
        const int xi1 = __builtin_bit_cast(int, xb);

        // B-matrix [h;x] 32x16 f16: lanes 0-15 hold K=0..15 (h0..15),
        // lanes 16-31 hold K=16..31 (x0..3 then zero pad).
        v8i bi;
        bi[0] = lo ? i0 : xi0;
        bi[1] = lo ? i1 : xi1;
        bi[2] = lo ? i2 : 0;
        bi[3] = lo ? i3 : 0;
        bi[4] = lo ? o0 : 0;
        bi[5] = lo ? o1 : 0;
        bi[6] = lo ? o2 : 0;
        bi[7] = lo ? o3 : 0;
        const v16h bm = __builtin_bit_cast(v16h, bi);

        // (neg_a, A, neg_b, B, c_mod, C, reuse_a, reuse_b)
        v8f gr  = __builtin_amdgcn_wmma_f32_16x16x32_f16(false, A_r,  false, bm,
                                                         (short)0, zc, false, false);
        v8f gz  = __builtin_amdgcn_wmma_f32_16x16x32_f16(false, A_z,  false, bm,
                                                         (short)0, zc, false, false);
        v8f gnh = __builtin_amdgcn_wmma_f32_16x16x32_f16(false, A_nh, false, bm,
                                                         (short)0, zc, false, false);
        v8f gnx = __builtin_amdgcn_wmma_f32_16x16x32_f16(false, A_nx, false, bm,
                                                         (short)0, zc, false, false);

        // Elementwise GRU update — layouts of g*, h all match (lane=batch,
        // VGPR j = hidden j+kOff), purely per-slot.
        #pragma unroll
        for (int j = 0; j < 8; ++j) {
            const float r  = sigf(gr[j] + br[j]);
            const float z  = sigf(gz[j] + bz[j]);
            const float nn = tanhf_fast(gnx[j] + bxn[j] + r * (gnh[j] + bhn[j]));
            h[j] = fmaf(z, h[j] - nn, nn);         // (1-z)*n + z*h
        }
    }

    // hT: batch = base+col, hidden = j + kOff (two aligned float4 stores)
    float* dst = hT_out + (size_t)(base + col) * H_RUNV + kOff;
    float4 s0 = {h[0], h[1], h[2], h[3]};
    float4 s1 = {h[4], h[5], h[6], h[7]};
    *(float4*)(dst)     = s0;
    *(float4*)(dst + 4) = s1;
}

// ---------------------------------------------------------------------------
// Kernel 2a: pre GRU input projection — embarrassingly parallel, streams the
// 120 MB `pre` tensor at full bandwidth. xg layout [B][T][3]: with
// idx = b*T + t, both reads (80 B/thread, contiguous across block) and writes
// (12 B/thread, contiguous across block) are coalesced.
// ---------------------------------------------------------------------------
__global__ __launch_bounds__(256)
void pre_inproj(const float* __restrict__ pre,    // [B, T, 20]
                const float* __restrict__ Wih_p,  // [3, 20]
                const float* __restrict__ bih_p,  // [3]
                float* __restrict__ xg) {         // [B, T, 3]
    const int idx = blockIdx.x * blockDim.x + threadIdx.x;
    if (idx >= B_TOT * T_STEPS) return;
    const float* xp = pre + (size_t)idx * 20;
    float xr = bih_p[0], xz = bih_p[1], xn = bih_p[2];
    #pragma unroll
    for (int i = 0; i < 20; ++i) {
        const float x = xp[i];
        xr = fmaf(Wih_p[i],      x, xr);
        xz = fmaf(Wih_p[20 + i], x, xz);
        xn = fmaf(Wih_p[40 + i], x, xn);
    }
    float* o = xg + (size_t)idx * 3;
    o[0] = xr; o[1] = xz; o[2] = xn;
}

// ---------------------------------------------------------------------------
// Kernel 2b: pre GRU scalar scan over precomputed xg (only 18 MB to stream,
// loads independent of the h-chain so they pipeline ahead).
// ---------------------------------------------------------------------------
__global__ __launch_bounds__(256)
void gru_pre_scan(const float* __restrict__ xg,     // [B, T, 3]
                  const float* __restrict__ Whh_p,  // [3, 1]
                  const float* __restrict__ bhh_p,  // [3]
                  float* __restrict__ out_pre) {    // [B, 16]
    const int b = blockIdx.x * blockDim.x + threadIdx.x;
    if (b >= B_TOT) return;
    const float whr = Whh_p[0], whz = Whh_p[1], whn = Whh_p[2];
    const float bhr = bhh_p[0], bhz = bhh_p[1], bhn = bhh_p[2];
    const float* g = xg + (size_t)b * (T_STEPS * 3);
    float h = 0.f;
    for (int t = 0; t < T_STEPS; ++t) {
        const int tp = (t + 48 < T_STEPS) ? (t + 48) : (T_STEPS - 1);
        __builtin_prefetch(g + tp * 3, 0, 1);
        const float xr = g[t * 3 + 0];
        const float xz = g[t * 3 + 1];
        const float xn = g[t * 3 + 2];
        const float r = sigf(xr + fmaf(whr, h, bhr));
        const float z = sigf(xz + fmaf(whz, h, bhz));
        const float n = tanhf_fast(xn + r * fmaf(whn, h, bhn));
        h = fmaf(z, h - n, n);
        const int s = t - (T_STEPS - 16);
        if (s >= 0) out_pre[(size_t)b * 16 + s] = h;
    }
}

// ---------------------------------------------------------------------------
// Kernel 2 (fallback, fused): used only if d_ws is too small for xg scratch.
// ---------------------------------------------------------------------------
__global__ __launch_bounds__(256)
void gru_pre_fused(const float* __restrict__ pre,
                   const float* __restrict__ Wih_p,
                   const float* __restrict__ Whh_p,
                   const float* __restrict__ bih_p,
                   const float* __restrict__ bhh_p,
                   float* __restrict__ out_pre) {
    const int b = blockIdx.x * blockDim.x + threadIdx.x;
    if (b >= B_TOT) return;
    float wr[20], wz[20], wn[20];
    #pragma unroll
    for (int i = 0; i < 20; ++i) {
        wr[i] = Wih_p[i];
        wz[i] = Wih_p[20 + i];
        wn[i] = Wih_p[40 + i];
    }
    const float whr = Whh_p[0], whz = Whh_p[1], whn = Whh_p[2];
    const float bir = bih_p[0], biz = bih_p[1], bin_ = bih_p[2];
    const float bhr = bhh_p[0], bhz = bhh_p[1], bhn = bhh_p[2];
    const float* xrow = pre + (size_t)b * (T_STEPS * 20);
    float h = 0.f;
    for (int t = 0; t < T_STEPS; ++t) {
        const float* xp = xrow + t * 20;
        float xr = bir, xz = biz, xn = bin_;
        #pragma unroll
        for (int i = 0; i < 20; ++i) {
            const float x = xp[i];
            xr = fmaf(wr[i], x, xr);
            xz = fmaf(wz[i], x, xz);
            xn = fmaf(wn[i], x, xn);
        }
        const float r = sigf(xr + fmaf(whr, h, bhr));
        const float z = sigf(xz + fmaf(whz, h, bhz));
        const float n = tanhf_fast(xn + r * fmaf(whn, h, bhn));
        h = fmaf(z, h - n, n);
        const int s = t - (T_STEPS - 16);
        if (s >= 0) out_pre[(size_t)b * 16 + s] = h;
    }
}

// ---------------------------------------------------------------------------
// Kernel 3: sigmoid(concat(h_runoff, out_pre) @ W_lin.T + b_lin)
// ---------------------------------------------------------------------------
__global__ __launch_bounds__(256)
void head_linear(const float* __restrict__ hro,    // [B,16]
                 const float* __restrict__ opre,   // [B,16]
                 const float* __restrict__ W_lin,  // [16,32]
                 const float* __restrict__ b_lin,  // [16]
                 float* __restrict__ out) {        // [B,16]
    const int idx = blockIdx.x * blockDim.x + threadIdx.x;
    const int b = idx >> 4;
    const int j = idx & 15;
    if (b >= B_TOT) return;
    const float* wj = W_lin + j * 32;
    float acc = b_lin[j];
    #pragma unroll
    for (int k = 0; k < 16; ++k) acc = fmaf(wj[k],      hro[(size_t)b * 16 + k], acc);
    #pragma unroll
    for (int k = 0; k < 16; ++k) acc = fmaf(wj[16 + k], opre[(size_t)b * 16 + k], acc);
    out[idx] = sigf(acc);
}

// ---------------------------------------------------------------------------
extern "C" void kernel_launch(void* const* d_in, const int* in_sizes, int n_in,
                              void* d_out, int out_size, void* d_ws, size_t ws_size,
                              hipStream_t stream) {
    const float* runoff = (const float*)d_in[0];
    const float* pre    = (const float*)d_in[1];
    const float* Wih_p  = (const float*)d_in[2];
    const float* Whh_p  = (const float*)d_in[3];
    const float* bih_p  = (const float*)d_in[4];
    const float* bhh_p  = (const float*)d_in[5];
    const float* Wih_r  = (const float*)d_in[6];
    const float* Whh_r  = (const float*)d_in[7];
    const float* bih_r  = (const float*)d_in[8];
    const float* bhh_r  = (const float*)d_in[9];
    const float* W_lin  = (const float*)d_in[10];
    const float* b_lin  = (const float*)d_in[11];

    float* out  = (float*)d_out;
    float* hro  = (float*)d_ws;                       // [B,16]
    float* opre = hro + (size_t)B_TOT * 16;           // [B,16]
    float* xg   = opre + (size_t)B_TOT * 16;          // [B,T,3] (optional)

    const size_t need = ((size_t)B_TOT * 16 * 2 + (size_t)B_TOT * T_STEPS * 3)
                        * sizeof(float);

    // 32 blocks x 8 waves x 16 samples = 4096
    gru_runoff_wmma<<<B_TOT / 128, 256, 0, stream>>>(runoff, Wih_r, Whh_r,
                                                     bih_r, bhh_r, hro);
    if (ws_size >= need) {
        const int n = B_TOT * T_STEPS;
        pre_inproj<<<(n + 255) / 256, 256, 0, stream>>>(pre, Wih_p, bih_p, xg);
        gru_pre_scan<<<B_TOT / 256, 256, 0, stream>>>(xg, Whh_p, bhh_p, opre);
    } else {
        gru_pre_fused<<<B_TOT / 256, 256, 0, stream>>>(pre, Wih_p, Whh_p,
                                                       bih_p, bhh_p, opre);
    }
    head_linear<<<(B_TOT * 16) / 256, 256, 0, stream>>>(hro, opre, W_lin, b_lin, out);
}
